// HET_RelationalAttLayer_21002390078097
// MI455X (gfx1250) — matrix operand order repaired and checked by reading, not compiled, not run on hardware
//
#include <hip/hip_runtime.h>
#include <hip/hip_bf16.h>
#include <stdint.h>

// ---------------- problem constants (from reference) ----------------
constexpr int N_NODES  = 50000;
constexpr int N_EDGES  = 819200;
constexpr int NUM_RELS = 8;
constexpr int N_HEADS  = 4;
constexpr int IN_FEAT  = 256;
constexpr int OUT_FEAT = 128;
constexpr int HEAD_DIM = OUT_FEAT / N_HEADS;   // 32
constexpr int EPR      = N_EDGES / NUM_RELS;   // 102400 edges per relation
constexpr float LEAKY  = 0.2f;

// ---------------- WMMA vector types (probe-confirmed signatures) ----------------
typedef __attribute__((ext_vector_type(16))) __bf16 v16bf;
typedef __attribute__((ext_vector_type(8)))  __bf16 v8bf;
typedef __attribute__((ext_vector_type(8)))  float  v8f;

// ---------------- helpers ----------------
__device__ __forceinline__ unsigned short f2bf(float f) {
    // round-to-nearest-even fp32 -> bf16
    unsigned u = __float_as_uint(f);
    unsigned r = u + 0x7fffu + ((u >> 16) & 1u);
    return (unsigned short)(r >> 16);
}

// monotone order-preserving float -> uint key (for atomicMax over signed floats)
__device__ __forceinline__ unsigned fkey(float f) {
    unsigned u = __float_as_uint(f);
    return (u & 0x80000000u) ? ~u : (u | 0x80000000u);
}
__device__ __forceinline__ float funkey(unsigned k) {
    unsigned u = (k & 0x80000000u) ? (k & 0x7fffffffu) : ~k;
    return __uint_as_float(u);
}
// fkey(-inf) == 0x007FFFFF -> init value for segment max
constexpr unsigned NEG_INF_KEY = 0x007FFFFFu;

// guaranteed-native fire-and-forget f32 L2 atomic (no return -> STOREcnt-tracked;
// s_endpgm's implicit wait-idle covers completion)
__device__ __forceinline__ void fatomic_add(float* p, float v) {
    asm volatile("global_atomic_add_f32 %0, %1, off"
                 :
                 : "v"(p), "v"(v)
                 : "memory");
}

__device__ __forceinline__ float leaky(float x) { return x > 0.f ? x : LEAKY * x; }

// ---------------- kernel 1a: inputs fp32 -> bf16 ----------------
__global__ void k_cvt_x(const float* __restrict__ x, unsigned short* __restrict__ xb) {
    int t = blockIdx.x * blockDim.x + threadIdx.x;
    if (t < N_NODES * IN_FEAT) xb[t] = f2bf(x[t]);
}

// ---------------- kernel 1b: conv_weights [R][H][K][D] -> Wt bf16 [R][n=h*32+d][K] ----
__global__ void k_cvt_w(const float* __restrict__ w, unsigned short* __restrict__ wt) {
    int t = blockIdx.x * blockDim.x + threadIdx.x;            // R*128*256 = 262144
    if (t >= NUM_RELS * OUT_FEAT * IN_FEAT) return;
    int k   = t & (IN_FEAT - 1);
    int n   = (t >> 8) & (OUT_FEAT - 1);
    int r   = t >> 15;
    int h   = n >> 5;
    int d   = n & 31;
    wt[t] = f2bf(w[(((size_t)r * N_HEADS + h) * IN_FEAT + k) * HEAD_DIM + d]);
}

// ---------------- kernel 2: relation GEMM, bf16 WMMA, f32 accumulate ----------------
// feat[r][node][n] = sum_k xb[node][k] * Wt[r][n][k]
// grid: NUM_RELS * (N_NODES/16) blocks of 128 threads = 4 waves.
// Each wave owns TWO adjacent 16-wide n-tiles: shared A fragment, two independent
// accumulator chains -> XDL co-execution instead of one serialized WMMA chain.
__global__ void __launch_bounds__(128) k_gemm_feat(
    const unsigned short* __restrict__ xb,   // [N_NODES][256] bf16
    const unsigned short* __restrict__ wt,   // [R][128][256]  bf16 (K contiguous)
    float* __restrict__ feat)                // [R][N_NODES][128] f32
{
    const int blk   = blockIdx.x;
    const int r     = blk / (N_NODES / 16);
    const int node0 = (blk % (N_NODES / 16)) * 16;
    const int wave  = threadIdx.x >> 5;      // 0..3 -> n-tile pair
    const int lane  = threadIdx.x & 31;
    const int n0    = wave * 32;             // first tile at n0, second at n0+16
    const int ml    = lane & 15;             // row-within-tile (A) / col (B,C,D)
    const int kh    = lane >> 4;             // K-half selector

    const unsigned short* xrow  = xb + (size_t)(node0 + ml) * IN_FEAT;
    const unsigned short* wrow0 = wt + ((size_t)r * OUT_FEAT + n0 + ml) * IN_FEAT;
    const unsigned short* wrow1 = wrow0 + (size_t)16 * IN_FEAT;

    v8f c0 = {}, c1 = {};
#pragma unroll
    for (int kb = 0; kb < IN_FEAT / 32; ++kb) {
        // A (16x32 bf16): lane half kh: elems j = K kb*32 + kh*16 + j (contiguous)
        const int ka = kb * 32 + kh * 16;
        v8bf alo = *(const v8bf*)(xrow + ka);
        v8bf ahi = *(const v8bf*)(xrow + ka + 8);
        v16bf a = __builtin_shufflevector(alo, ahi, 0,1,2,3,4,5,6,7,8,9,10,11,12,13,14,15);
        // B (32x16 bf16): lane = col n, elems j = K kb*32 + kh*16 + j (contiguous in Wt)
        v16bf b0 = *(const v16bf*)(wrow0 + kb * 32 + kh * 16);
        v16bf b1 = *(const v16bf*)(wrow1 + kb * 32 + kh * 16);
        c0 = __builtin_amdgcn_wmma_f32_16x16x32_bf16(false, a, false, b0, (short)0, c0,
                                                     false, false);
        c1 = __builtin_amdgcn_wmma_f32_16x16x32_bf16(false, a, false, b1, (short)0, c1,
                                                     false, false);
    }
    // D (16x16 f32): vgpr i -> M = kh*8 + i, N = ml
    float* o = feat + ((size_t)r * N_NODES + node0 + kh * 8) * OUT_FEAT + n0 + ml;
#pragma unroll
    for (int i = 0; i < 8; ++i) {
        o[(size_t)i * OUT_FEAT]      = c0[i];
        o[(size_t)i * OUT_FEAT + 16] = c1[i];
    }
}

// ---------------- kernel 3: per-(r,node,head) attention dot products ----------------
__global__ void k_attn_dots(const float* __restrict__ feat,
                            const float* __restrict__ attn_l,
                            const float* __restrict__ attn_r,
                            float* __restrict__ el, float* __restrict__ er)
{
    int t = blockIdx.x * blockDim.x + threadIdx.x;           // R*N*H = 1.6M
    if (t >= NUM_RELS * N_NODES * N_HEADS) return;
    int h = t & (N_HEADS - 1);
    int n = (t >> 2) % N_NODES;
    int r = t / (N_NODES * N_HEADS);
    const float* f  = feat + ((size_t)r * N_NODES + n) * OUT_FEAT + h * HEAD_DIM;
    const float* al = attn_l + (r * N_HEADS + h) * HEAD_DIM;
    const float* ar = attn_r + (r * N_HEADS + h) * HEAD_DIM;
    float sl = 0.f, sr = 0.f;
#pragma unroll
    for (int d = 0; d < HEAD_DIM; ++d) { float v = f[d]; sl += v * al[d]; sr += v * ar[d]; }
    el[t] = sl;
    er[t] = sr;
}

// ---------------- kernel 4: init accumulators (runs every call; ws/out are poisoned) --
__global__ void k_init(float* __restrict__ out, unsigned* __restrict__ mkey,
                       float* __restrict__ denom)
{
    int t = blockIdx.x * blockDim.x + threadIdx.x;
    if (t < N_NODES * OUT_FEAT) out[t] = 0.f;
    if (t < N_NODES * N_HEADS) { mkey[t] = NEG_INF_KEY; denom[t] = 0.f; }
}

// ---------------- kernel 5: segment max of edge scores ----------------
__global__ void k_edge_max(const int* __restrict__ row_idx, const int* __restrict__ col_idx,
                           const float* __restrict__ el, const float* __restrict__ er,
                           unsigned* __restrict__ mkey)
{
    int t = blockIdx.x * blockDim.x + threadIdx.x;           // E*H
    if (t >= N_EDGES * N_HEADS) return;
    int h = t & (N_HEADS - 1);
    int e = t >> 2;
    int r = e / EPR;
    int row = row_idx[e], col = col_idx[e];
    float s = leaky(el[((size_t)r * N_NODES + row) * N_HEADS + h] +
                    er[((size_t)r * N_NODES + col) * N_HEADS + h]);
    atomicMax(&mkey[col * N_HEADS + h], fkey(s));
}

// ---------------- kernel 6: exp + segment sum ----------------
__global__ void k_edge_exp(const int* __restrict__ row_idx, const int* __restrict__ col_idx,
                           const float* __restrict__ el, const float* __restrict__ er,
                           const unsigned* __restrict__ mkey,
                           float* __restrict__ eexp, float* __restrict__ denom)
{
    int t = blockIdx.x * blockDim.x + threadIdx.x;           // E*H
    if (t >= N_EDGES * N_HEADS) return;
    int h = t & (N_HEADS - 1);
    int e = t >> 2;
    int r = e / EPR;
    int row = row_idx[e], col = col_idx[e];
    float s = leaky(el[((size_t)r * N_NODES + row) * N_HEADS + h] +
                    er[((size_t)r * N_NODES + col) * N_HEADS + h]);
    float m  = funkey(mkey[col * N_HEADS + h]);
    float ee = __expf(s - m);
    eexp[t] = ee;
    fatomic_add(&denom[col * N_HEADS + h], ee);
}

// ---------------- kernel 7: weighted scatter of features ----------------
// one wave per edge; lane owns 4 contiguous output features (all in one head)
__global__ void __launch_bounds__(256) k_edge_scatter(
    const int* __restrict__ row_idx, const int* __restrict__ col_idx,
    const float* __restrict__ eexp, const float* __restrict__ denom,
    const float* __restrict__ feat, float* __restrict__ out)
{
    int e = blockIdx.x * 8 + (threadIdx.x >> 5);
    if (e >= N_EDGES) return;
    int lane = threadIdx.x & 31;
    int r = e / EPR;
    int row = row_idx[e], col = col_idx[e];
    int h = lane >> 3;                                       // HEAD_DIM/4 = 8 lanes per head
    float alpha = eexp[(size_t)e * N_HEADS + h] / denom[col * N_HEADS + h];
    const float4 f = *(const float4*)(feat + ((size_t)r * N_NODES + row) * OUT_FEAT + lane * 4);
    float* o = out + (size_t)col * OUT_FEAT + lane * 4;
    fatomic_add(o + 0, alpha * f.x);
    fatomic_add(o + 1, alpha * f.y);
    fatomic_add(o + 2, alpha * f.z);
    fatomic_add(o + 3, alpha * f.w);
}

// ---------------- kernel 8: add bias ----------------
__global__ void k_bias(float* __restrict__ out, const float* __restrict__ bias) {
    int t = blockIdx.x * blockDim.x + threadIdx.x;
    if (t < N_NODES * OUT_FEAT) out[t] += bias[t & (OUT_FEAT - 1)];
}

// ---------------- launch ----------------
extern "C" void kernel_launch(void* const* d_in, const int* in_sizes, int n_in,
                              void* d_out, int out_size, void* d_ws, size_t ws_size,
                              hipStream_t stream)
{
    const float* x       = (const float*)d_in[0];   // [N, 256]
    const float* convw   = (const float*)d_in[1];   // [8,4,256,32]
    const float* attn_l  = (const float*)d_in[2];   // [8,4,32]
    const float* attn_r  = (const float*)d_in[3];   // [8,4,32]
    const float* h_bias  = (const float*)d_in[4];   // [128]
    const int*   row_idx = (const int*)d_in[5];     // [E]
    const int*   col_idx = (const int*)d_in[6];     // [E]
    float*       out     = (float*)d_out;           // [N, 128]
    (void)in_sizes; (void)n_in; (void)out_size;

    // workspace carve-up (~258.5 MB total), 256B aligned
    char* ws = (char*)d_ws;
    size_t off = 0;
    auto carve = [&](size_t bytes) -> char* {
        char* p = ws + off;
        off = (off + bytes + 255) & ~(size_t)255;
        return p;
    };
    unsigned short* xb    = (unsigned short*)carve((size_t)N_NODES * IN_FEAT * 2);            // 25.6 MB
    unsigned short* wt    = (unsigned short*)carve((size_t)NUM_RELS * OUT_FEAT * IN_FEAT * 2);// 0.5 MB
    float*          feat  = (float*)carve((size_t)NUM_RELS * N_NODES * OUT_FEAT * 4);         // 204.8 MB
    float*          el    = (float*)carve((size_t)NUM_RELS * N_NODES * N_HEADS * 4);          // 6.4 MB
    float*          er    = (float*)carve((size_t)NUM_RELS * N_NODES * N_HEADS * 4);          // 6.4 MB
    unsigned*       mkey  = (unsigned*)carve((size_t)N_NODES * N_HEADS * 4);                  // 0.8 MB
    float*          denom = (float*)carve((size_t)N_NODES * N_HEADS * 4);                     // 0.8 MB
    float*          eexp  = (float*)carve((size_t)N_EDGES * N_HEADS * 4);                     // 13.1 MB
    (void)ws_size; // requires ~260 MB of scratch

    constexpr int TB = 256;
    // 1. precision staging
    k_cvt_x<<<(N_NODES * IN_FEAT + TB - 1) / TB, TB, 0, stream>>>(x, xb);
    k_cvt_w<<<(NUM_RELS * OUT_FEAT * IN_FEAT + TB - 1) / TB, TB, 0, stream>>>(convw, wt);
    // 2. relation-grouped GEMM on WMMA (8 * [50000x256]@[256x128], f32 accum)
    k_gemm_feat<<<NUM_RELS * (N_NODES / 16), 128, 0, stream>>>(xb, wt, feat);
    // 3. per-(r,node,head) attention projections
    k_attn_dots<<<(NUM_RELS * N_NODES * N_HEADS + TB - 1) / TB, TB, 0, stream>>>(
        feat, attn_l, attn_r, el, er);
    // 4. init out / segment-max keys / denominators
    k_init<<<(N_NODES * OUT_FEAT + TB - 1) / TB, TB, 0, stream>>>(out, mkey, denom);
    // 5-6. edge softmax (segment max, then exp + segment sum)
    k_edge_max<<<(N_EDGES * N_HEADS + TB - 1) / TB, TB, 0, stream>>>(
        row_idx, col_idx, el, er, mkey);
    k_edge_exp<<<(N_EDGES * N_HEADS + TB - 1) / TB, TB, 0, stream>>>(
        row_idx, col_idx, el, er, mkey, eexp, denom);
    // 7. alpha-weighted feature scatter (one wave per edge)
    k_edge_scatter<<<N_EDGES / 8, TB, 0, stream>>>(row_idx, col_idx, eexp, denom, feat, out);
    // 8. bias
    k_bias<<<(N_NODES * OUT_FEAT + TB - 1) / TB, TB, 0, stream>>>(out, h_bias);
}
